// GraphFeatureEncoder_61280593379655
// MI455X (gfx1250) — compile-verified, matrix operand order checked by reading, not run on hardware
//
#include <hip/hip_runtime.h>
#include <hip/hip_bf16.h>

// ---------------------------------------------------------------------------
// GraphFeatureEncoder for MI455X (gfx1250, wave32, WMMA + TDM)
//
//   per layer:
//     1. Wh = f16(W)
//     2. P  = xh @ Wh^T   [N, H*Cout] f16 via v_wmma_f32_16x16x32_f16
//        - A tile DMA'd into LDS by the Tensor Data Mover (tensor_load_to_lds)
//        - D tile staged in LDS -> coalesced b128 stores
//     3. Q  = xh @ U^T    [N, 8]      f32 (VALU, tiny)
//     4. edge pass: 8-head softmax + gather P[src] (51MB, L2-resident in 192MB)
//        -> atomicAdd into agg[dst]
//     5. self-loop pass (attn = softmax(c))
//     6. 1/deg + bias + relu + BN stats (ds_add_f32 -> global atomics)
//     7. fold BN -> next-layer xh (f16)
// ---------------------------------------------------------------------------

typedef __attribute__((ext_vector_type(16))) _Float16     v16h;
typedef __attribute__((ext_vector_type(8)))  _Float16     v8h;
typedef __attribute__((ext_vector_type(8)))  float        v8f;
typedef __attribute__((ext_vector_type(4)))  unsigned int v4u;
typedef __attribute__((ext_vector_type(8)))  int          v8i;
typedef __attribute__((ext_vector_type(4)))  int          v4i;

#define NNODES 50000
#define NEDGES 800000
#define NHEADS 8

// ---------------------------------------------------------------- TDM helper
// 2-D tile load Global->LDS via Tensor Data Mover (CDNA5 §8 D# descriptor).
// rows x cols f16 elements, row stride in halves. Issued once per wave;
// completion via s_wait_tensorcnt. This toolchain's builtin takes 6 args
// (clang-23 / therock-10.0 variant): g0, g1, g2, g3, extra, cpol.
__device__ __forceinline__ void tdm_load_tile_f16(unsigned lds_off, const void* gptr,
                                                  int rows, int cols, int stride) {
  unsigned long long ga = (unsigned long long)(uintptr_t)gptr;
  v4u g0;
  g0.x = 0x1u;                                    // count=1, user descriptor
  g0.y = lds_off;                                 // D#.lds_addr (bytes)
  g0.z = (unsigned)(ga & 0xFFFFFFFFull);          // global_addr[31:0]
  g0.w = (unsigned)((ga >> 32) & 0x01FFFFFFull)   // global_addr[56:32]
         | (2u << 30);                            // type=2 ("image")
  v8i g1;
  g1[0] = (1 << 16);                              // wg_mask=0, data_size=1 (2B)
  g1[1] = (cols & 0xFFFF) << 16;                  // tensor_dim0[15:0]
  g1[2] = ((unsigned)cols >> 16) | ((rows & 0xFFFF) << 16);  // dim0 hi | dim1 lo
  g1[3] = (((unsigned)rows >> 16) & 0xFFFF) | ((cols & 0xFFFF) << 16); // dim1 hi | tile_dim0
  g1[4] = rows & 0xFFFF;                          // tile_dim1 (tile_dim2=0)
  g1[5] = stride;                                 // tensor_dim0_stride[31:0]
  g1[6] = 0;
  g1[7] = 0;
  v4i z4 = {0, 0, 0, 0};                          // groups 2/3 unused (2-D tensor)
  v8i z8 = {0, 0, 0, 0, 0, 0, 0, 0};
  __builtin_amdgcn_tensor_load_to_lds(g0, g1, z4, z4, z8, 0);
}

// ---------------------------------------------------------------- elementwise
__global__ __launch_bounds__(256) void k_fill(float* p, int n, float v) {
  int i = blockIdx.x * 256 + threadIdx.x;
  if (i < n) p[i] = v;
}

__global__ __launch_bounds__(256) void k_recip(float* p, int n) {
  int i = blockIdx.x * 256 + threadIdx.x;
  if (i < n) p[i] = 1.0f / p[i];
}

__global__ __launch_bounds__(256) void k_cvt_f32_f16(const float* __restrict__ in,
                                                     _Float16* __restrict__ out, int n) {
  int i = blockIdx.x * 256 + threadIdx.x;
  if (i < n) out[i] = (_Float16)in[i];
}

__global__ __launch_bounds__(256) void k_count_acc(const int* __restrict__ dst,
                                                   float* cnt, int e) {
  int i = blockIdx.x * 256 + threadIdx.x;
  if (i < e) atomicAdd(&cnt[dst[i]], 1.0f);
}

// ---------------------------------------------------------------- P = xh @ Wh^T
// A: xh [N, 64] f16 row-major (TDM -> LDS).  B: Wh [MOUT, 64] f16 row-major.
// One wave computes a 16x16 tile of P (2x K=32 WMMA); block = 4 waves = 16x64.
// D staged in LDS then written out as coalesced 16B stores.
__global__ __launch_bounds__(128)
void k_gemm_p(const _Float16* __restrict__ xh, const _Float16* __restrict__ Wh,
              _Float16* __restrict__ P, int MOUT) {
  __shared__ __align__(16) _Float16 atile[16 * 64];
  __shared__ __align__(16) _Float16 dtile[16 * 64];

  const int lane = threadIdx.x;            // 0..31
  const int wave = threadIdx.y;            // 0..3
  const int row0 = blockIdx.x * 16;        // node-row tile
  const int colb = blockIdx.y * 64;        // block column base
  const int col0 = colb + wave * 16;       // this wave's column tile

  // A tile: 16 rows x 64 halves, DMA'd by the Tensor Data Mover (wave 0 issues;
  // flat shared addr[31:0] is the LDS byte offset per the aperture mapping).
  if (wave == 0) {
    tdm_load_tile_f16((unsigned)(uintptr_t)atile, xh + (size_t)row0 * 64, 16, 64, 64);
    __builtin_amdgcn_s_wait_tensorcnt(0);
  }
  __syncthreads();

  const int m  = lane & 15;                // A: row / B: column within tile
  const int kh = (lane >> 4) * 8;          // K half-offset (0 or 8)

  // A fragments from LDS, per 7.12.2 16-bit A layout
  v8h a00 = *(const v8h*)(atile + m * 64 + kh);
  v8h a01 = *(const v8h*)(atile + m * 64 + kh + 16);
  v8h a10 = *(const v8h*)(atile + m * 64 + 32 + kh);
  v8h a11 = *(const v8h*)(atile + m * 64 + 32 + kh + 16);
  v16h A0 = __builtin_shufflevector(a00, a01, 0,1,2,3,4,5,6,7,8,9,10,11,12,13,14,15);
  v16h A1 = __builtin_shufflevector(a10, a11, 0,1,2,3,4,5,6,7,8,9,10,11,12,13,14,15);

  // B fragments from global (W tiny, cache-resident)
  const _Float16* wrow = Wh + (size_t)(col0 + m) * 64;
  v8h b00 = *(const v8h*)(wrow + kh);
  v8h b01 = *(const v8h*)(wrow + kh + 16);
  v8h b10 = *(const v8h*)(wrow + 32 + kh);
  v8h b11 = *(const v8h*)(wrow + 32 + kh + 16);
  v16h B0 = __builtin_shufflevector(b00, b01, 0,1,2,3,4,5,6,7,8,9,10,11,12,13,14,15);
  v16h B1 = __builtin_shufflevector(b10, b11, 0,1,2,3,4,5,6,7,8,9,10,11,12,13,14,15);

  v8f acc = {};
  acc = __builtin_amdgcn_wmma_f32_16x16x32_f16(false, A0, false, B0, (short)0, acc, false, false);
  acc = __builtin_amdgcn_wmma_f32_16x16x32_f16(false, A1, false, B1, (short)0, acc, false, false);

  // D layout: element r -> (M = r + 8*(lane/16), N = lane&15); stage in LDS.
  const int mb = 8 * (lane >> 4);
#pragma unroll
  for (int r = 0; r < 8; ++r)
    dtile[(mb + r) * 64 + wave * 16 + m] = (_Float16)acc[r];
  __syncthreads();

  // coalesced write-out: 128 threads x one 16B chunk
  {
    const int t = wave * 32 + lane;        // 0..127
    const int r = t >> 3;                  // row 0..15
    const int ch = (t & 7) * 8;            // half-column chunk
    *(v8h*)(P + (size_t)(row0 + r) * MOUT + colb + ch) = *(const v8h*)(dtile + r * 64 + ch);
  }
}

// ---------------------------------------------------------------- Q = xh @ U^T
__global__ __launch_bounds__(256)
void k_q(const _Float16* __restrict__ xh, const float* __restrict__ U,
         float* __restrict__ Q, int total /* N*8 */) {
  int idx = blockIdx.x * 256 + threadIdx.x;
  if (idx >= total) return;
  int n = idx >> 3, h = idx & 7;
  const _Float16* xr = xh + (size_t)n * 64;
  const float*    ur = U + h * 64;
  float acc = 0.f;
#pragma unroll
  for (int k = 0; k < 64; ++k) acc += (float)xr[k] * ur[k];
  Q[idx] = acc;
}

// ---------------------------------------------------------------- edge pass
template <int COUT>
__global__ __launch_bounds__(256)
void k_edge(const int* __restrict__ src, const int* __restrict__ dst,
            const float* __restrict__ Q, const _Float16* __restrict__ P,
            const float* __restrict__ cvec, float* __restrict__ agg, int e_cnt) {
  int e = blockIdx.x * 8 + threadIdx.y;
  if (e >= e_cnt) return;
  const int s = src[e], d = dst[e];

  const _Float16* prow = P + (size_t)s * (NHEADS * COUT);
  __builtin_prefetch(prow, 0, 3);  // global_prefetch

  float l[NHEADS], mx = -1e30f;
#pragma unroll
  for (int h = 0; h < NHEADS; ++h) {
    l[h] = Q[d * NHEADS + h] - Q[s * NHEADS + h] + cvec[h];
    mx = fmaxf(mx, l[h]);
  }
  float sum = 0.f;
#pragma unroll
  for (int h = 0; h < NHEADS; ++h) { l[h] = __expf(l[h] - mx); sum += l[h]; }
  const float inv = 1.0f / sum;

  const int lane = threadIdx.x;
#pragma unroll
  for (int i = 0; i < COUT / 32; ++i) {
    const int c = lane + 32 * i;
    float acc = 0.f;
#pragma unroll
    for (int h = 0; h < NHEADS; ++h) acc += l[h] * (float)prow[h * COUT + c];
    atomicAdd(&agg[(size_t)d * COUT + c], acc * inv);
  }
}

// ---------------------------------------------------------------- self loops
template <int COUT>
__global__ __launch_bounds__(256)
void k_self(const float* __restrict__ cvec, const _Float16* __restrict__ P,
            float* __restrict__ agg, int n_nodes) {
  int n = blockIdx.x * 8 + threadIdx.y;
  if (n >= n_nodes) return;
  float l[NHEADS], mx = -1e30f;
#pragma unroll
  for (int h = 0; h < NHEADS; ++h) { l[h] = cvec[h]; mx = fmaxf(mx, l[h]); }
  float sum = 0.f;
#pragma unroll
  for (int h = 0; h < NHEADS; ++h) { l[h] = __expf(l[h] - mx); sum += l[h]; }
  const float inv = 1.0f / sum;

  const _Float16* prow = P + (size_t)n * (NHEADS * COUT);
  const int lane = threadIdx.x;
#pragma unroll
  for (int i = 0; i < COUT / 32; ++i) {
    const int c = lane + 32 * i;
    float acc = 0.f;
#pragma unroll
    for (int h = 0; h < NHEADS; ++h) acc += l[h] * (float)prow[h * COUT + c];
    agg[(size_t)n * COUT + c] += acc * inv;
  }
}

// ------------------------------------------- 1/deg + bias + relu + BN stats
__global__ __launch_bounds__(256)
void k_post_stats(float* __restrict__ agg, const float* __restrict__ invc,
                  const float* __restrict__ bias, float* __restrict__ bnsum,
                  float* __restrict__ bnsq, int total /* N*64 */) {
  __shared__ float ssum[64], ssq[64];
  const int tid = threadIdx.x;
  if (tid < 64) { ssum[tid] = 0.f; ssq[tid] = 0.f; }
  __syncthreads();
  const int idx = blockIdx.x * 256 + tid;
  if (idx < total) {
    const int n = idx >> 6, c = idx & 63;
    float y = fmaxf(agg[idx] * invc[n] + bias[c], 0.f);
    agg[idx] = y;
    atomicAdd(&ssum[c], y);        // ds_add_f32
    atomicAdd(&ssq[c], y * y);
  }
  __syncthreads();
  if (tid < 64) {
    atomicAdd(&bnsum[tid], ssum[tid]);
    atomicAdd(&bnsq[tid], ssq[tid]);
  }
}

__global__ void k_bnfold(const float* bnsum, const float* bnsq, const float* g,
                         const float* be, float* scale, float* shift, float invN) {
  const int c = threadIdx.x;  // 64 threads
  const float m = bnsum[c] * invN;
  const float v = bnsq[c] * invN - m * m;
  const float s = g[c] * rsqrtf(v + 1e-5f);
  scale[c] = s;
  shift[c] = be[c] - m * s;
}

__global__ __launch_bounds__(256)
void k_apply_bn(const float* __restrict__ agg, const float* __restrict__ scale,
                const float* __restrict__ shift, _Float16* __restrict__ xh, int total) {
  const int idx = blockIdx.x * 256 + threadIdx.x;
  if (idx < total) {
    const int c = idx & 63;
    xh[idx] = (_Float16)(agg[idx] * scale[c] + shift[c]);
  }
}

__global__ __launch_bounds__(256)
void k_out(const float* __restrict__ agg, const float* __restrict__ invc,
           const float* __restrict__ bias, float* __restrict__ out, int total /* N*32 */) {
  const int idx = blockIdx.x * 256 + threadIdx.x;
  if (idx < total) {
    const int n = idx >> 5, c = idx & 31;
    out[idx] = agg[idx] * invc[n] + bias[c];
  }
}

// ---------------------------------------------------------------------------
extern "C" void kernel_launch(void* const* d_in, const int* in_sizes, int n_in,
                              void* d_out, int out_size, void* d_ws, size_t ws_size,
                              hipStream_t stream) {
  (void)in_sizes; (void)n_in; (void)out_size; (void)ws_size;
  const int N = NNODES, E = NEDGES;

  const float* x   = (const float*)d_in[0];
  const int*   ei  = (const int*)d_in[1];
  const int*   src = ei;
  const int*   dst = ei + E;
  const float* W1  = (const float*)d_in[2];
  const float* U1  = (const float*)d_in[3];
  const float* c1  = (const float*)d_in[4];
  const float* b1  = (const float*)d_in[5];
  const float* g1  = (const float*)d_in[6];
  const float* be1 = (const float*)d_in[7];
  const float* W2  = (const float*)d_in[8];
  const float* U2  = (const float*)d_in[9];
  const float* c2  = (const float*)d_in[10];
  const float* b2  = (const float*)d_in[11];
  const float* g2  = (const float*)d_in[12];
  const float* be2 = (const float*)d_in[13];
  const float* W3  = (const float*)d_in[14];
  const float* U3  = (const float*)d_in[15];
  const float* c3  = (const float*)d_in[16];
  const float* b3  = (const float*)d_in[17];

  // workspace partition (all chunks 16B-aligned by construction)
  char* w = (char*)d_ws;
  _Float16* xh  = (_Float16*)w; w += (size_t)N * 64 * 2;    // 6.4 MB
  _Float16* P   = (_Float16*)w; w += (size_t)N * 512 * 2;   // 51.2 MB (L2-resident)
  float* Q      = (float*)w;    w += (size_t)N * 8 * 4;     // 1.6 MB
  float* agg    = (float*)w;    w += (size_t)N * 64 * 4;    // 12.8 MB
  float* cnt    = (float*)w;    w += (size_t)N * 4;         // 0.2 MB
  _Float16* Wh  = (_Float16*)w; w += (size_t)512 * 64 * 2;  // 64 KB
  float* bnsum  = (float*)w;    w += 64 * 4;
  float* bnsq   = (float*)w;    w += 64 * 4;
  float* bnscale= (float*)w;    w += 64 * 4;
  float* bnshift= (float*)w;    w += 64 * 4;

  const int B256 = 256;
  const int nb_N64 = (N * 64 + B256 - 1) / B256;
  const int nb_N32 = (N * 32 + B256 - 1) / B256;
  const int nb_N8  = (N * 8 + B256 - 1) / B256;
  const int nb_N   = (N + B256 - 1) / B256;
  const int nb_E   = (E + B256 - 1) / B256;

  // ---- degree counts (self-loop => init 1) ----
  k_fill<<<nb_N, B256, 0, stream>>>(cnt, N, 1.0f);
  k_count_acc<<<nb_E, B256, 0, stream>>>(dst, cnt, E);
  k_recip<<<nb_N, B256, 0, stream>>>(cnt, N);

  // ---- xh = f16(x) ----
  k_cvt_f32_f16<<<nb_N64, B256, 0, stream>>>(x, xh, N * 64);

  // ================= layer 1 (64 -> 64, BN) =================
  k_cvt_f32_f16<<<(512 * 64) / B256, B256, 0, stream>>>(W1, Wh, 512 * 64);
  k_gemm_p<<<dim3(N / 16, 8), dim3(32, 4), 0, stream>>>(xh, Wh, P, 512);
  k_q<<<nb_N8, B256, 0, stream>>>(xh, U1, Q, N * 8);
  k_fill<<<nb_N64, B256, 0, stream>>>(agg, N * 64, 0.0f);
  k_edge<64><<<(E + 7) / 8, dim3(32, 8), 0, stream>>>(src, dst, Q, P, c1, agg, E);
  k_self<64><<<(N + 7) / 8, dim3(32, 8), 0, stream>>>(c1, P, agg, N);
  k_fill<<<1, B256, 0, stream>>>(bnsum, 64, 0.0f);
  k_fill<<<1, B256, 0, stream>>>(bnsq, 64, 0.0f);
  k_post_stats<<<nb_N64, B256, 0, stream>>>(agg, cnt, b1, bnsum, bnsq, N * 64);
  k_bnfold<<<1, 64, 0, stream>>>(bnsum, bnsq, g1, be1, bnscale, bnshift, 1.0f / N);
  k_apply_bn<<<nb_N64, B256, 0, stream>>>(agg, bnscale, bnshift, xh, N * 64);

  // ================= layer 2 (64 -> 64, BN) =================
  k_cvt_f32_f16<<<(512 * 64) / B256, B256, 0, stream>>>(W2, Wh, 512 * 64);
  k_gemm_p<<<dim3(N / 16, 8), dim3(32, 4), 0, stream>>>(xh, Wh, P, 512);
  k_q<<<nb_N8, B256, 0, stream>>>(xh, U2, Q, N * 8);
  k_fill<<<nb_N64, B256, 0, stream>>>(agg, N * 64, 0.0f);
  k_edge<64><<<(E + 7) / 8, dim3(32, 8), 0, stream>>>(src, dst, Q, P, c2, agg, E);
  k_self<64><<<(N + 7) / 8, dim3(32, 8), 0, stream>>>(c2, P, agg, N);
  k_fill<<<1, B256, 0, stream>>>(bnsum, 64, 0.0f);
  k_fill<<<1, B256, 0, stream>>>(bnsq, 64, 0.0f);
  k_post_stats<<<nb_N64, B256, 0, stream>>>(agg, cnt, b2, bnsum, bnsq, N * 64);
  k_bnfold<<<1, 64, 0, stream>>>(bnsum, bnsq, g2, be2, bnscale, bnshift, 1.0f / N);
  k_apply_bn<<<nb_N64, B256, 0, stream>>>(agg, bnscale, bnshift, xh, N * 64);

  // ================= layer 3 (64 -> 32, no BN/relu) =================
  k_cvt_f32_f16<<<(256 * 64) / B256, B256, 0, stream>>>(W3, Wh, 256 * 64);
  k_gemm_p<<<dim3(N / 16, 4), dim3(32, 4), 0, stream>>>(xh, Wh, P, 256);
  k_q<<<nb_N8, B256, 0, stream>>>(xh, U3, Q, N * 8);
  k_fill<<<nb_N32, B256, 0, stream>>>(agg, N * 32, 0.0f);
  k_edge<32><<<(E + 7) / 8, dim3(32, 8), 0, stream>>>(src, dst, Q, P, c3, agg, E);
  k_self<32><<<(N + 7) / 8, dim3(32, 8), 0, stream>>>(c3, P, agg, N);
  k_out<<<nb_N32, B256, 0, stream>>>(agg, cnt, b3, (float*)d_out, N * 32);
}
